// SwitchableLayerNorm_82008105550601
// MI455X (gfx1250) — compile-verified, matrix operand order checked by reading, not run on hardware
//
#include <hip/hip_runtime.h>
#include <hip/hip_bf16.h>

typedef __attribute__((ext_vector_type(16))) _Float16 v16h;
typedef __attribute__((ext_vector_type(8)))  float    v8f;
typedef __attribute__((ext_vector_type(4)))  int      v4i;

#define DDIM 768
#define KBUK 4
#define EPS_F 1e-5f

// ---------- CDNA5 async global->LDS staging (ASYNCcnt path) ----------
__device__ __forceinline__ void async_copy_b128(float* lds_dst, const float* gsrc) {
#if __has_builtin(__builtin_amdgcn_global_load_async_to_lds_b128)
  typedef __attribute__((address_space(1))) float gfloat_t;
  typedef __attribute__((address_space(3))) float lfloat_t;
  gfloat_t* g1 = (gfloat_t*)const_cast<float*>(gsrc);     // addrspacecast generic->global
  lfloat_t* l1 = (lfloat_t*)lds_dst;                      // addrspacecast generic->LDS
  __builtin_amdgcn_global_load_async_to_lds_b128(
      (__attribute__((address_space(1))) v4i*)g1,
      (__attribute__((address_space(3))) v4i*)l1,
      0, 0);
#else
  unsigned lds_off = (unsigned)(unsigned long long)lds_dst;
  asm volatile("global_load_async_to_lds_b128 %0, %1, off"
               :: "v"(lds_off), "v"(gsrc) : "memory");
#endif
}

__device__ __forceinline__ void wait_asynccnt0() {
#if __has_builtin(__builtin_amdgcn_s_wait_asynccnt)
  __builtin_amdgcn_s_wait_asynccnt(0);
#else
  asm volatile("s_wait_asynccnt 0" ::: "memory");
#endif
}

__device__ __forceinline__ float readlane_f(float v, int lane) {
  return __builtin_bit_cast(float,
      __builtin_amdgcn_readlane(__builtin_bit_cast(int, v), lane));
}

// ---------------- main kernel: 8 waves/block, 16 rows per wave ----------------
__global__ __launch_bounds__(256) void sln_main(
    const float* __restrict__ x, const float* __restrict__ weights,
    const float* __restrict__ biases, const int* __restrict__ bucket,
    float* __restrict__ out, float* __restrict__ ws)
{
  __shared__ __align__(16) float s_wb[2 * KBUK * DDIM];  // 24 KB: weights | biases
  __shared__ float s_part[8][12];                        // per-wave {sumMean[4], sumVar[4], cnt[4]}

  const int tid  = threadIdx.x;
  const int lane = tid & 31;
  const int wave = tid >> 5;

  // stage affine tables into LDS asynchronously (hidden behind pass 1)
#pragma unroll
  for (int t = 0; t < 3; ++t) {
    const int idx = t * 1024 + tid * 4;              // 256 thr * 4 floats = 1024/round
    async_copy_b128(&s_wb[idx],                weights + idx);
    async_copy_b128(&s_wb[KBUK * DDIM + idx],  biases  + idx);
  }

  const int tile = blockIdx.x * 8 + wave;
  const int row0 = tile * 16;
  const int r    = lane & 15;
  const int half = lane >> 4;                        // A-fragment half selector

  const float* xr = x + (size_t)(row0 + r) * DDIM;

  // ---- pass 1: row sums & sums of squares via WMMA with B = ones ----
  v8f cS = {}; v8f cQ = {};
  v16h onesB;
#pragma unroll
  for (int i = 0; i < 16; ++i) onesB[i] = (_Float16)1.0f;

  for (int c = 0; c < DDIM; c += 32) {
    const int k1 = c + (half ? 8 : 0);               // K 0-7  or 8-15
    const int k2 = c + 16 + (half ? 8 : 0);          // K16-23 or 24-31
    float4 f0 = *(const float4*)(xr + k1);
    float4 f1 = *(const float4*)(xr + k1 + 4);
    float4 f2 = *(const float4*)(xr + k2);
    float4 f3 = *(const float4*)(xr + k2 + 4);
    float fv[16] = {f0.x,f0.y,f0.z,f0.w, f1.x,f1.y,f1.z,f1.w,
                    f2.x,f2.y,f2.z,f2.w, f3.x,f3.y,f3.z,f3.w};
    v16h a, q;
#pragma unroll
    for (int i = 0; i < 16; ++i) {
      a[i] = (_Float16)fv[i];
      q[i] = (_Float16)(fv[i] * fv[i]);
    }
    cS = __builtin_amdgcn_wmma_f32_16x16x32_f16(false, a, false, onesB, (short)0, cS, false, false);
    cQ = __builtin_amdgcn_wmma_f32_16x16x32_f16(false, q, false, onesB, (short)0, cQ, false, false);
  }

  const int vb = bucket[row0 + r];                   // lanes 0..15 hold this tile's buckets

  wait_asynccnt0();                                  // wb tables now resident
  __syncthreads();

  // ---- pass 2 (fused): stats extraction, bucket partials, normalize+affine ----
  float sm[KBUK] = {0,0,0,0}, sv[KBUK] = {0,0,0,0};
  int   cnt[KBUK] = {0,0,0,0};
  const float invD = 1.0f / (float)DDIM;

#pragma unroll
  for (int m = 0; m < 16; ++m) {
    const int src = (m < 8) ? 0 : 16;
    const float S = readlane_f(cS[m & 7], src);
    const float Q = readlane_f(cQ[m & 7], src);
    const int  kb = __builtin_amdgcn_readlane(vb, m);

    const float mean = S * invD;
    const float var  = fmaf(-mean, mean, Q * invD);
    const float rstd = rsqrtf(var + EPS_F);

#pragma unroll
    for (int k = 0; k < KBUK; ++k) {
      const bool h = (kb == k);
      sm[k]  += h ? mean : 0.0f;
      sv[k]  += h ? var  : 0.0f;
      cnt[k] += h ? 1 : 0;
    }

    const float* xr2 = x   + (size_t)(row0 + m) * DDIM;
    float*       or2 = out + (size_t)(row0 + m) * DDIM;
    const float* wr  = s_wb + kb * DDIM;
    const float* br  = s_wb + KBUK * DDIM + kb * DDIM;
    const float nms  = -mean * rstd;

#pragma unroll
    for (int j = 0; j < 6; ++j) {
      const int col = j * 128 + lane * 4;
      float4 v  = *(const float4*)(xr2 + col);
      float4 w  = *(const float4*)(wr + col);
      float4 bb = *(const float4*)(br + col);
      float4 o;
      o.x = fmaf(fmaf(v.x, rstd, nms), w.x, bb.x);
      o.y = fmaf(fmaf(v.y, rstd, nms), w.y, bb.y);
      o.z = fmaf(fmaf(v.z, rstd, nms), w.z, bb.z);
      o.w = fmaf(fmaf(v.w, rstd, nms), w.w, bb.w);
      *(float4*)(or2 + col) = o;
    }
  }

  if (lane == 0) {
#pragma unroll
    for (int k = 0; k < KBUK; ++k) {
      s_part[wave][k]     = sm[k];
      s_part[wave][4 + k] = sv[k];
      s_part[wave][8 + k] = (float)cnt[k];
    }
  }
  __syncthreads();

  if (tid < KBUK) {                                  // fixed-order wave reduction -> deterministic
    float a = 0.f, b = 0.f, c2 = 0.f;
#pragma unroll
    for (int w8 = 0; w8 < 8; ++w8) {
      a  += s_part[w8][tid];
      b  += s_part[w8][4 + tid];
      c2 += s_part[w8][8 + tid];
    }
    float* slot = ws + (size_t)blockIdx.x * 12;      // written, never accumulated: no init needed
    slot[tid]     = a;
    slot[4 + tid] = b;
    slot[8 + tid] = c2;
  }
}

// ---------------- finalize: deterministic block-order reduction + EMA ----------------
__global__ void sln_finalize(const float* __restrict__ ws, int nblk,
                             const float* __restrict__ buckets,
                             const int* __restrict__ amounts,
                             float* __restrict__ outTail, float invN)
{
  const int k = threadIdx.x;
  if (k >= KBUK) return;
  float smean = 0.f, svar = 0.f, cn = 0.f;
  for (int b = 0; b < nblk; ++b) {
    smean += ws[b * 12 + k];
    svar  += ws[b * 12 + 4 + k];
    cn    += ws[b * 12 + 8 + k];
  }
  const int added  = (int)(cn + 0.5f);
  const int newAmt = amounts[k] + added;
  const bool nz    = (newAmt != 0);
  const float rel  = nz ? (float)added / (float)(newAmt > 1 ? newAmt : 1) : 0.0f;
  const float mb   = smean * invN;
  const float vb   = svar  * invN;
  const float b0   = buckets[2 * k], b1 = buckets[2 * k + 1];
  outTail[2 * k]     = nz ? mb * rel + b0 * (1.0f - rel) : b0;
  outTail[2 * k + 1] = nz ? vb * rel + b1 * (1.0f - rel) : b1;
  outTail[8 + k]     = (float)newAmt;                // int32 result emitted as value-cast float
}

extern "C" void kernel_launch(void* const* d_in, const int* in_sizes, int n_in,
                              void* d_out, int out_size, void* d_ws, size_t ws_size,
                              hipStream_t stream) {
  const float* x       = (const float*)d_in[0];
  const float* weights = (const float*)d_in[1];
  const float* biases  = (const float*)d_in[2];
  const float* buckets = (const float*)d_in[3];
  const int*   amounts = (const int*)d_in[4];
  const int*   bucket  = (const int*)d_in[5];
  float*       out     = (float*)d_out;
  float*       wsf     = (float*)d_ws;

  const int N      = in_sizes[5];        // 65536
  const int blocks = N / 128;            // 8 waves * 16 rows per block

  sln_main<<<blocks, 256, 0, stream>>>(x, weights, biases, bucket, out, wsf);
  sln_finalize<<<1, 32, 0, stream>>>(wsf, blocks, buckets, amounts,
                                     out + (size_t)N * DDIM, 1.0f / (float)N);
}
